// DINModel_49589692400164
// MI455X (gfx1250) — matrix operand order repaired and checked by reading, not compiled, not run on hardware
//
#include <hip/hip_runtime.h>
#include <hip/hip_bf16.h>
#include <math.h>

typedef __attribute__((ext_vector_type(16))) _Float16 v16h;
typedef __attribute__((ext_vector_type(8)))  _Float16 v8h;
typedef __attribute__((ext_vector_type(4)))  _Float16 v4h;
typedef __attribute__((ext_vector_type(8)))  float    v8f;

#define WMMA_F32_F16(a, b, c) \
  __builtin_amdgcn_wmma_f32_16x16x32_f16(false, (a), false, (b), (short)0, (c), false, false)

// Load 16 f16 values as two 16B chunks at p and p+16 halves.
// Used both for A fragments (lo half-K block, hi half-K block are 16 halves apart)
// and for B fragments (the same two 8-half chunks from a transposed weight image).
__device__ __forceinline__ v16h frag16(const _Float16* p) {
  v8h lo = *(const v8h*)p;
  v8h hi = *(const v8h*)(p + 16);
  v16h r;
#pragma unroll
  for (int i = 0; i < 8; ++i) { r[i] = lo[i]; r[8 + i] = hi[i]; }
  return r;
}

__device__ __forceinline__ v4h cvt4(float4 v) {
  v4h h;
  h[0] = (_Float16)v.x; h[1] = (_Float16)v.y;
  h[2] = (_Float16)v.z; h[3] = (_Float16)v.w;
  return h;
}

// ---------------------------------------------------------------------------
// Kernel A: per-batch-row fused attention (gather + 3-layer MLP + masked
// softmax + weighted sum + context MLP). One workgroup (256 thr = 8 waves)
// per b. 13 row-tiles of 16 (l = 0..207, rows >=200 are zero-padded).
// ---------------------------------------------------------------------------
__global__ __launch_bounds__(256)
void din_attention_kernel(const int* __restrict__ iseq,    // (B,200)
                          const int* __restrict__ slen_arr,// (B,1)
                          const int* __restrict__ titem,   // (B,1)
                          const float* __restrict__ ctxf,  // (B,9)
                          const float* __restrict__ emb,   // (100001,64)
                          const float* __restrict__ aw1,   // (128,64)
                          const float* __restrict__ ab1,   // (64)
                          const float* __restrict__ aw2,   // (64,32)
                          const float* __restrict__ ab2,   // (32)
                          const float* __restrict__ aw3,   // (32,1)
                          const float* __restrict__ ab3,   // (1)
                          const float* __restrict__ cw,    // (9,32)
                          const float* __restrict__ cb,    // (32)
                          _Float16* __restrict__ comb)     // (B,160) f16 out
{
  __shared__ __align__(32) _Float16 s_aw1T[64][128];  // aw1 transposed [N][K], 16 KB
  __shared__ __align__(32) _Float16 s_aw2T[32][64];   // aw2 transposed [N][K], 4 KB
  __shared__ __align__(32) _Float16 s_seq[201][64];   // gathered seq emb (row 200 = zeros)
  __shared__ __align__(32) _Float16 s_h[8][16][64];   // per-wave h1/h2 restage
  __shared__ __align__(32) _Float16 s_tgt[64];
  __shared__ float s_aw3v[32];
  __shared__ float s_ab1v[64];
  __shared__ float s_ab2v[32];
  __shared__ float s_ctx[32];
  __shared__ float s_scores[200];
  __shared__ float s_red[256];

  const int b    = blockIdx.x;
  const int tid  = threadIdx.x;
  const int lane = tid & 31;
  const int wave = tid >> 5;
  const int m    = lane & 15;   // row within 16-row tile / column index for B frag
  const int hs   = lane >> 4;   // half-select
  const int slen = slen_arr[b];

  // --- stage weights (transposed to [N][K] so B fragments are contiguous) ---
  for (int e = tid * 4; e < 128 * 64; e += 1024) {
    float4 w = *(const float4*)(aw1 + e);
    int k = e >> 6, n = e & 63;
    s_aw1T[n + 0][k] = (_Float16)w.x;
    s_aw1T[n + 1][k] = (_Float16)w.y;
    s_aw1T[n + 2][k] = (_Float16)w.z;
    s_aw1T[n + 3][k] = (_Float16)w.w;
  }
  for (int e = tid * 4; e < 64 * 32; e += 1024) {
    float4 w = *(const float4*)(aw2 + e);
    int k = e >> 5, n = e & 31;
    s_aw2T[n + 0][k] = (_Float16)w.x;
    s_aw2T[n + 1][k] = (_Float16)w.y;
    s_aw2T[n + 2][k] = (_Float16)w.z;
    s_aw2T[n + 3][k] = (_Float16)w.w;
  }
  if (tid < 32) s_aw3v[tid] = aw3[tid];
  if (tid < 64) s_ab1v[tid] = ab1[tid];
  if (tid < 32) s_ab2v[tid] = ab2[tid];
  if (tid < 16) {
    float4 v = *(const float4*)(emb + (long)titem[b] * 64 + tid * 4);
    *(v4h*)&s_tgt[tid * 4] = cvt4(v);
  }
  if (tid < 32) {
    float acc = cb[tid];
#pragma unroll
    for (int j = 0; j < 9; ++j) acc += ctxf[b * 9 + j] * cw[j * 32 + tid];
    s_ctx[tid] = fmaxf(acc, 0.f);
  }
  // --- gather 200 sequence embedding rows into LDS (f16), row 200 = zeros ---
  for (int p = tid; p < 201 * 16; p += 256) {
    int r = p >> 4, c4 = (p & 15) << 2;
    float4 v = {0.f, 0.f, 0.f, 0.f};
    if (r < 200) v = *(const float4*)(emb + (long)iseq[b * 200 + r] * 64 + c4);
    *(v4h*)&s_seq[r][c4] = cvt4(v);
  }
  __syncthreads();

  const float ab3v = ab3[0];

  // --- per-wave attention MLP over 16-row tiles of l ---
  for (int t = wave; t < 13; t += 8) {
    // layer 1: (16 x 128) @ (128 x 64): 4 K-chunks x 4 N-tiles of WMMA
    v8f acc[4] = {};
#pragma unroll
    for (int c = 0; c < 4; ++c) {
      const _Float16* ap;
      if (c < 2) {
        ap = &s_tgt[c * 32 + 8 * hs];              // query half (same for all rows)
      } else {
        int l = t * 16 + m;
        if (l > 200) l = 200;                       // padded rows -> zero row
        ap = &s_seq[l][(c - 2) * 32 + 8 * hs];
      }
      v16h a = frag16(ap);
#pragma unroll
      for (int nt = 0; nt < 4; ++nt) {
        v16h bf = frag16(&s_aw1T[nt * 16 + m][c * 32 + 8 * hs]);
        acc[nt] = WMMA_F32_F16(a, bf, acc[nt]);
      }
    }
    // bias + relu, restage D-layout -> row-major f16 in per-wave LDS tile
#pragma unroll
    for (int nt = 0; nt < 4; ++nt) {
      int n2 = nt * 16 + m;
      float bia = s_ab1v[n2];
#pragma unroll
      for (int v = 0; v < 8; ++v)
        s_h[wave][v + 8 * hs][n2] = (_Float16)fmaxf(acc[nt][v] + bia, 0.f);
    }
    asm volatile("s_wait_dscnt 0" ::: "memory");

    // layer 2: (16 x 64) @ (64 x 32): 2 K-chunks x 2 N-tiles
    v8f acc2[2] = {};
#pragma unroll
    for (int c = 0; c < 2; ++c) {
      v16h a = frag16(&s_h[wave][m][c * 32 + 8 * hs]);
#pragma unroll
      for (int nt = 0; nt < 2; ++nt) {
        v16h bf = frag16(&s_aw2T[nt * 16 + m][c * 32 + 8 * hs]);
        acc2[nt] = WMMA_F32_F16(a, bf, acc2[nt]);
      }
    }
#pragma unroll
    for (int nt = 0; nt < 2; ++nt) {
      int n2 = nt * 16 + m;
      float bia = s_ab2v[n2];
#pragma unroll
      for (int v = 0; v < 8; ++v)
        s_h[wave][v + 8 * hs][n2] = (_Float16)fmaxf(acc2[nt][v] + bia, 0.f);
    }
    asm volatile("s_wait_dscnt 0" ::: "memory");

    // layer 3 (32 -> 1): one row per lane 0..15, masked write to scores
    if (hs == 0) {
      float s = ab3v;
#pragma unroll
      for (int k = 0; k < 32; ++k) s += (float)s_h[wave][m][k] * s_aw3v[k];
      int l = t * 16 + m;
      if (l < 200) s_scores[l] = (l < slen) ? s : -1.0e9f;
    }
  }
  __syncthreads();

  // --- masked softmax over 200 positions ---
  float sv = (tid < 200) ? s_scores[tid] : -3.0e38f;
  s_red[tid] = sv;
  __syncthreads();
  for (int off = 128; off > 0; off >>= 1) {
    if (tid < off) s_red[tid] = fmaxf(s_red[tid], s_red[tid + off]);
    __syncthreads();
  }
  float mx = s_red[0];
  __syncthreads();
  float ex = (tid < 200) ? __expf(sv - mx) : 0.f;
  s_red[tid] = ex;
  __syncthreads();
  for (int off = 128; off > 0; off >>= 1) {
    if (tid < off) s_red[tid] += s_red[tid + off];
    __syncthreads();
  }
  float inv = 1.f / s_red[0];
  __syncthreads();
  if (tid < 200) s_scores[tid] = ex * inv;
  __syncthreads();

  // --- attended = sum_l w_l * seq_emb[l][d] (4 partial stripes x 64 dims) ---
  {
    int d = tid & 63, g = tid >> 6;
    float acc = 0.f;
    for (int l = g; l < 200; l += 4) acc += s_scores[l] * (float)s_seq[l][d];
    s_red[g * 64 + d] = acc;
  }
  __syncthreads();

  // --- write fused comb row [attended(64) | tgt(64) | ctx(32)] as f16 ---
  _Float16* crow = comb + (long)b * 160;
  if (tid < 64) {
    crow[tid] = (_Float16)(s_red[tid] + s_red[64 + tid] + s_red[128 + tid] + s_red[192 + tid]);
  } else if (tid < 128) {
    crow[tid] = s_tgt[tid - 64];
  } else if (tid < 160) {
    crow[tid] = (_Float16)s_ctx[tid - 128];
  }
}

// ---------------------------------------------------------------------------
// Kernel B1: main MLP layer 1: (8192 x 160) @ (160 x 128) + relu -> h1 (f16)
// 64 blocks x 8 waves; each wave one 16-row M-tile. 5 K-chunks x 8 N-tiles.
// ---------------------------------------------------------------------------
__global__ __launch_bounds__(256)
void din_mlp1_kernel(const _Float16* __restrict__ comb,
                     const float* __restrict__ mw1, const float* __restrict__ mb1,
                     _Float16* __restrict__ h1ws)
{
  __shared__ __align__(32) _Float16 s_w[128][160];  // mw1 transposed [N][K], 40 KB
  __shared__ float s_b[128];
  const int tid = threadIdx.x, lane = tid & 31, wave = tid >> 5;
  const int m = lane & 15, hs = lane >> 4;

  for (int e = tid * 4; e < 160 * 128; e += 1024) {
    float4 w = *(const float4*)(mw1 + e);
    int k = e >> 7, n = e & 127;
    s_w[n + 0][k] = (_Float16)w.x;
    s_w[n + 1][k] = (_Float16)w.y;
    s_w[n + 2][k] = (_Float16)w.z;
    s_w[n + 3][k] = (_Float16)w.w;
  }
  if (tid < 128) s_b[tid] = mb1[tid];
  __syncthreads();

  const int rt = blockIdx.x * 8 + wave;  // 512 M-tiles
  const long r0 = (long)rt * 16;
  const _Float16* arow = comb + (r0 + m) * 160;

  v8f acc[8] = {};
#pragma unroll
  for (int c = 0; c < 5; ++c) {
    v16h a = frag16(arow + c * 32 + 8 * hs);
#pragma unroll
    for (int nt = 0; nt < 8; ++nt) {
      v16h bf = frag16(&s_w[nt * 16 + m][c * 32 + 8 * hs]);
      acc[nt] = WMMA_F32_F16(a, bf, acc[nt]);
    }
  }
#pragma unroll
  for (int nt = 0; nt < 8; ++nt) {
    int n2 = nt * 16 + m;
    float bia = s_b[n2];
#pragma unroll
    for (int v = 0; v < 8; ++v)
      h1ws[(r0 + v + 8 * hs) * 128 + n2] = (_Float16)fmaxf(acc[nt][v] + bia, 0.f);
  }
}

// ---------------------------------------------------------------------------
// Kernel B2: layer 2 (128->64) + relu + layer 3 (64->1) + sigmoid -> out
// ---------------------------------------------------------------------------
__global__ __launch_bounds__(256)
void din_mlp2_kernel(const _Float16* __restrict__ h1ws,
                     const float* __restrict__ mw2, const float* __restrict__ mb2,
                     const float* __restrict__ mw3, const float* __restrict__ mb3,
                     float* __restrict__ out)
{
  __shared__ __align__(32) _Float16 s_w[64][128];       // mw2 transposed, 16 KB
  __shared__ __align__(32) _Float16 s_h2[8][16][64];    // per-wave h2, 16 KB
  __shared__ float s_b[64];
  __shared__ float s_w3[64];
  const int tid = threadIdx.x, lane = tid & 31, wave = tid >> 5;
  const int m = lane & 15, hs = lane >> 4;

  for (int e = tid * 4; e < 128 * 64; e += 1024) {
    float4 w = *(const float4*)(mw2 + e);
    int k = e >> 6, n = e & 63;
    s_w[n + 0][k] = (_Float16)w.x;
    s_w[n + 1][k] = (_Float16)w.y;
    s_w[n + 2][k] = (_Float16)w.z;
    s_w[n + 3][k] = (_Float16)w.w;
  }
  if (tid < 64) { s_b[tid] = mb2[tid]; s_w3[tid] = mw3[tid]; }
  __syncthreads();

  const int rt = blockIdx.x * 8 + wave;
  const long r0 = (long)rt * 16;
  const _Float16* arow = h1ws + (r0 + m) * 128;

  v8f acc[4] = {};
#pragma unroll
  for (int c = 0; c < 4; ++c) {
    v16h a = frag16(arow + c * 32 + 8 * hs);
#pragma unroll
    for (int nt = 0; nt < 4; ++nt) {
      v16h bf = frag16(&s_w[nt * 16 + m][c * 32 + 8 * hs]);
      acc[nt] = WMMA_F32_F16(a, bf, acc[nt]);
    }
  }
#pragma unroll
  for (int nt = 0; nt < 4; ++nt) {
    int n2 = nt * 16 + m;
    float bia = s_b[n2];
#pragma unroll
    for (int v = 0; v < 8; ++v)
      s_h2[wave][v + 8 * hs][n2] = (_Float16)fmaxf(acc[nt][v] + bia, 0.f);
  }
  asm volatile("s_wait_dscnt 0" ::: "memory");

  if (hs == 0) {
    float x = mb3[0];
#pragma unroll
    for (int k = 0; k < 64; ++k) x += (float)s_h2[wave][m][k] * s_w3[k];
    out[r0 + m] = 1.f / (1.f + __expf(-x));
  }
}

// ---------------------------------------------------------------------------
extern "C" void kernel_launch(void* const* d_in, const int* in_sizes, int n_in,
                              void* d_out, int out_size, void* d_ws, size_t ws_size,
                              hipStream_t stream) {
  const int*   iseq  = (const int*)  d_in[0];   // item_sequence
  // d_in[1] click_sequence: unused by reference
  const int*   slen  = (const int*)  d_in[2];   // seq_length
  const int*   titem = (const int*)  d_in[3];   // target_item
  const float* ctxf  = (const float*)d_in[4];   // context_features
  const float* emb   = (const float*)d_in[5];   // emb_table
  const float* aw1   = (const float*)d_in[6];
  const float* ab1   = (const float*)d_in[7];
  const float* aw2   = (const float*)d_in[8];
  const float* ab2   = (const float*)d_in[9];
  const float* aw3   = (const float*)d_in[10];
  const float* ab3   = (const float*)d_in[11];
  const float* cw    = (const float*)d_in[12];
  const float* cb    = (const float*)d_in[13];
  const float* mw1   = (const float*)d_in[14];
  const float* mb1   = (const float*)d_in[15];
  const float* mw2   = (const float*)d_in[16];
  const float* mb2   = (const float*)d_in[17];
  const float* mw3   = (const float*)d_in[18];
  const float* mb3   = (const float*)d_in[19];
  float* out = (float*)d_out;

  const int B = 8192;
  _Float16* comb = (_Float16*)d_ws;                                // B*160 f16
  _Float16* h1   = (_Float16*)((char*)d_ws + (size_t)B * 160 * 2); // B*128 f16

  din_attention_kernel<<<B, 256, 0, stream>>>(iseq, slen, titem, ctxf, emb,
                                              aw1, ab1, aw2, ab2, aw3, ab3,
                                              cw, cb, comb);
  din_mlp1_kernel<<<B / 16 / 8, 256, 0, stream>>>(comb, mw1, mb1, h1);
  din_mlp2_kernel<<<B / 16 / 8, 256, 0, stream>>>(h1, mw2, mb2, mw3, mb3, out);
}
